// MaskedMultiHeadAttention_21388937134305
// MI455X (gfx1250) — matrix operand order repaired and checked by reading, not compiled, not run on hardware
//
#include <hip/hip_runtime.h>
#include <hip/hip_bf16.h>
#include <stdint.h>

// ---------------- CDNA5 WMMA types ----------------
typedef __attribute__((ext_vector_type(16))) __bf16       bf16x16;
typedef __attribute__((ext_vector_type(8)))  float        f32x8;
typedef __attribute__((ext_vector_type(8)))  unsigned int u32x8;

#define B_  4
#define S_  2048
#define D_  1024
#define H_  16
#define DK_ 64
#define DV_ 64

// ---------------- helpers ----------------
__device__ __forceinline__ unsigned short f32_bf16(float f) {
  union { float f; unsigned u; } v; v.f = f;
  unsigned r = 0x7FFFu + ((v.u >> 16) & 1u);   // round-to-nearest-even
  return (unsigned short)((v.u + r) >> 16);
}
__device__ __forceinline__ unsigned pack2(float a, float b) {
  return (unsigned)f32_bf16(a) | ((unsigned)f32_bf16(b) << 16);
}

// Build a 16xbf16 fragment from two 16B-aligned 8-element chunks.
__device__ __forceinline__ bf16x16 frag2(const unsigned short* p0, const unsigned short* p1) {
  uint4 a = *(const uint4*)p0;
  uint4 b = *(const uint4*)p1;
  u32x8 u;
  u[0] = a.x; u[1] = a.y; u[2] = a.z; u[3] = a.w;
  u[4] = b.x; u[5] = b.y; u[6] = b.z; u[7] = b.w;
  return __builtin_bit_cast(bf16x16, u);
}
// 16 contiguous bf16.
__device__ __forceinline__ bf16x16 frag1(const unsigned short* p) { return frag2(p, p + 8); }

__device__ __forceinline__ f32x8 wmma_bf16(bf16x16 a, bf16x16 b, f32x8 c) {
  // (neg_a, A, neg_b, B, c_mod, C, reuse_a, reuse_b)
  return __builtin_amdgcn_wmma_f32_16x16x32_bf16(false, a, false, b, (short)0, c, false, false);
}

// Cross-lane reductions inside each 16-lane half (wave32 C/D layout).
__device__ __forceinline__ float hred_max(float v) {
#pragma unroll
  for (int m = 1; m < 16; m <<= 1) v = fmaxf(v, __shfl_xor(v, m, 32));
  return v;
}
__device__ __forceinline__ float hred_sum(float v) {
#pragma unroll
  for (int m = 1; m < 16; m <<= 1) v += __shfl_xor(v, m, 32);
  return v;
}

// ---- cooperative 16x64 GEMM tile: acc[4] += A[16x1024] * BtHead[64x1024]^T.
// 8 waves/WG share BtHead; B k-slices (64x32 bf16 = 4KB) are double-buffered
// in LDS via global_load_async_to_lds_b128 (each wave moves a 512B slice).
__device__ __forceinline__ void coop_gemm16x64(const unsigned short* __restrict__ A,
                                               const unsigned short* __restrict__ BtHead,
                                               unsigned short (*ldsB)[64][32],
                                               int wv, int lane, f32x8 acc[4]) {
  const int l15 = lane & 15, lhi = lane >> 4;
  const unsigned short* ar = A + l15 * D_ + lhi * 8;

  // this wave's async slice: rows wv*8 .. wv*8+7, 64B per row per k-step
  const int srow = wv * 8 + (lane >> 2);
  const int schunk = (lane & 3) * 8;                    // element offset (16B chunks)
  const unsigned short* gsrc = BtHead + (size_t)srow * D_ + schunk;
  unsigned lds0 = (unsigned)(uintptr_t)&ldsB[0][srow][schunk];
  unsigned lds1 = (unsigned)(uintptr_t)&ldsB[1][srow][schunk];

  asm volatile("global_load_async_to_lds_b128 %0, %1, off"
               :: "v"(lds0), "v"((unsigned long long)(uintptr_t)gsrc) : "memory");

  for (int j = 0; j < D_ / 32; ++j) {
    asm volatile("s_wait_asynccnt 0" ::: "memory");     // own slice landed
    __syncthreads();                                    // all slices landed; prev reads done
    if (j + 1 < D_ / 32) {
      unsigned ldst = (j & 1) ? lds0 : lds1;
      asm volatile("global_load_async_to_lds_b128 %0, %1, off"
                   :: "v"(ldst),
                      "v"((unsigned long long)(uintptr_t)(gsrc + (size_t)(j + 1) * 32))
                   : "memory");
    }
    const int k0 = j * 32;
    bf16x16 af = frag2(ar + k0, ar + k0 + 16);
    const unsigned short (*bufp)[32] = ldsB[j & 1];
#pragma unroll
    for (int t = 0; t < 4; ++t) {
      bf16x16 bf = frag2(&bufp[t * 16 + l15][lhi * 8], &bufp[t * 16 + l15][lhi * 8 + 16]);
      acc[t] = wmma_bf16(af, bf, acc[t]);
    }
  }
  __syncthreads();
}

// ---------------- conversion kernels ----------------
__global__ __launch_bounds__(256) void conv_x(const float* __restrict__ x,
                                              unsigned short* __restrict__ xb, int n4) {
  int i = blockIdx.x * 256 + threadIdx.x;
  if (i < n4) {
    float4 f = ((const float4*)x)[i];
    uint2 o; o.x = pack2(f.x, f.y); o.y = pack2(f.z, f.w);
    ((uint2*)xb)[i] = o;
  }
}

// Wt[((sel*16+h)*64 + n)*1024 + d] = W_sel[h][d][n]  (bf16, transposed per head)
__global__ __launch_bounds__(256) void conv_wqkv(const float* __restrict__ Wq,
                                                 const float* __restrict__ Wk,
                                                 const float* __restrict__ Wv,
                                                 unsigned short* __restrict__ Wt) {
  int i = blockIdx.x * 256 + threadIdx.x;
  if (i < 3 * H_ * DK_ * D_) {
    int d = i & 1023, n = (i >> 10) & 63, h = (i >> 16) & 15, sel = i >> 20;
    const float* W = (sel == 0) ? Wq : (sel == 1) ? Wk : Wv;
    Wt[i] = f32_bf16(W[(h * D_ + d) * DK_ + n]);
  }
}

// Wot[n*1024 + k] = Wo[k*1024 + n]  (bf16, transposed)
__global__ __launch_bounds__(256) void conv_wo(const float* __restrict__ Wo,
                                               unsigned short* __restrict__ Wot) {
  int i = blockIdx.x * 256 + threadIdx.x;
  if (i < D_ * D_) {
    int k = i & 1023, n = i >> 10;
    Wot[i] = f32_bf16(Wo[k * D_ + n]);
  }
}

// ---------------- QKV projection (16x64 tile per wave) ----------------
// wave id bits: [8:0]=m-tile(512) [14:9]=(sel*16+h) -> 8 waves/WG share head
__global__ __launch_bounds__(256) void qkv_gemm(const unsigned short* __restrict__ xb,
                                                const unsigned short* __restrict__ Wt,
                                                unsigned short* __restrict__ qb,
                                                unsigned short* __restrict__ kb,
                                                unsigned short* __restrict__ vT) {
  __shared__ __align__(16) unsigned short ldsB[2][64][32];   // 8KB  B k-slices
  __shared__ __align__(16) unsigned short stg[8][16][64];    // 16KB store bounce

  int wv = threadIdx.x >> 5, lane = threadIdx.x & 31;
  int gw = blockIdx.x * 8 + wv;
  int mt = gw & 511, hs = gw >> 9;
  int h = hs & 15, sel = hs >> 4;

  f32x8 acc[4];
#pragma unroll
  for (int t = 0; t < 4; ++t) acc[t] = (f32x8){0.f, 0.f, 0.f, 0.f, 0.f, 0.f, 0.f, 0.f};

  coop_gemm16x64(xb + (size_t)(mt * 16) * D_,
                 Wt + (size_t)(sel * H_ + h) * DK_ * D_,
                 ldsB, wv, lane, acc);

  int l15 = lane & 15, lhi = lane >> 4;
  int m0 = mt * 16;
  int b = m0 >> 11, s0 = m0 & (S_ - 1);

  if (sel == 2) {
    // vT[b][h][dv][s]: per lane col dv fixed, rows r are 8 consecutive s.
#pragma unroll
    for (int t = 0; t < 4; ++t) {
      unsigned short tv[8];
#pragma unroll
      for (int r = 0; r < 8; ++r) tv[r] = f32_bf16(acc[t][r]);
      unsigned short* dst =
          vT + ((size_t)((b * H_ + h) * DV_ + t * 16 + l15)) * S_ + s0 + lhi * 8;
      uint4 pk;
      pk.x = (unsigned)tv[0] | ((unsigned)tv[1] << 16);
      pk.y = (unsigned)tv[2] | ((unsigned)tv[3] << 16);
      pk.z = (unsigned)tv[4] | ((unsigned)tv[5] << 16);
      pk.w = (unsigned)tv[6] | ((unsigned)tv[7] << 16);
      *(uint4*)dst = pk;
    }
  } else {
    // D-layout -> LDS -> coalesced b128 stores of [s][dk] rows.
#pragma unroll
    for (int t = 0; t < 4; ++t)
#pragma unroll
      for (int r = 0; r < 8; ++r)
        stg[wv][r + 8 * lhi][t * 16 + l15] = f32_bf16(acc[t][r]);
    asm volatile("s_wait_dscnt 0" ::: "memory");
    unsigned short* rowbase =
        ((sel == 0) ? qb : kb) + ((size_t)((b * H_ + h) * S_ + s0)) * DK_;
#pragma unroll
    for (int it = 0; it < 4; ++it) {
      int chunk = it * 32 + lane;
      int row = chunk >> 3, c8 = (chunk & 7) * 8;
      uint4 vld = *(const uint4*)&stg[wv][row][c8];
      *(uint4*)(rowbase + (size_t)row * DK_ + c8) = vld;
    }
  }
}

// ---------------- flash attention (causal) ----------------
// wave id bits: [6:0]=q-tile(128) [10:7]=head [12:11]=batch
__global__ __launch_bounds__(256) void attn(const unsigned short* __restrict__ qb,
                                            const unsigned short* __restrict__ kb,
                                            const unsigned short* __restrict__ vT,
                                            unsigned short* __restrict__ ctxb) {
  __shared__ __align__(16) unsigned short pT[8][16][32];  // per-wave P tile (bf16)

  int wv = threadIdx.x >> 5, lane = threadIdx.x & 31;
  int gw = blockIdx.x * 8 + wv;
  int qt = gw & 127, h = (gw >> 7) & 15, b = gw >> 11;
  int q0 = qt * 16;
  int l15 = lane & 15, lhi = lane >> 4;

  const unsigned short* qrow  = qb + ((size_t)((b * H_ + h) * S_ + q0 + l15)) * DK_;
  const unsigned short* kbase = kb + ((size_t)((b * H_ + h) * S_)) * DK_;
  const unsigned short* vbase = vT + ((size_t)((b * H_ + h) * DV_)) * S_;

  bf16x16 qf0 = frag2(qrow + lhi * 8,      qrow + lhi * 8 + 16);
  bf16x16 qf1 = frag2(qrow + 32 + lhi * 8, qrow + 32 + lhi * 8 + 16);

  float mst[8], lst[8];
  f32x8 cacc[4];
#pragma unroll
  for (int r = 0; r < 8; ++r) { mst[r] = -3.0e38f; lst[r] = 0.f; }
#pragma unroll
  for (int t = 0; t < 4; ++t) cacc[t] = (f32x8){0.f, 0.f, 0.f, 0.f, 0.f, 0.f, 0.f, 0.f};

  const float scale = 0.125f;  // 1/sqrt(64)
  int nblk = (q0 + 15) / 32 + 1;

  for (int blk = 0; blk < nblk; ++blk) {
    int kv0 = blk * 32;
    if (blk + 1 < nblk) {  // -> global_prefetch_b8
      __builtin_prefetch(kbase + (size_t)(kv0 + 32 + l15) * DK_, 0, 0);
      __builtin_prefetch(vbase + (size_t)l15 * S_ + kv0 + 32, 0, 0);
    }

    // scores: two 16-key subtiles, each K=64 (2 WMMAs)
    f32x8 st[2];
#pragma unroll
    for (int sub = 0; sub < 2; ++sub) {
      const unsigned short* krow = kbase + (size_t)(kv0 + sub * 16 + l15) * DK_;
      bf16x16 b0 = frag2(krow + lhi * 8, krow + lhi * 8 + 16);
      bf16x16 b1 = frag2(krow + 32 + lhi * 8, krow + 32 + lhi * 8 + 16);
      f32x8 s = {0.f, 0.f, 0.f, 0.f, 0.f, 0.f, 0.f, 0.f};
      s = wmma_bf16(qf0, b0, s);
      s = wmma_bf16(qf1, b1, s);
      st[sub] = s;
    }

    // causal mask + scale + online softmax
    float p0[8], p1[8], alpha[8];
#pragma unroll
    for (int r = 0; r < 8; ++r) {
      int qg  = q0 + r + 8 * lhi;
      int kg0 = kv0 + l15;
      int kg1 = kv0 + 16 + l15;
      float sc0 = (kg0 <= qg) ? st[0][r] * scale : -1.0e30f;
      float sc1 = (kg1 <= qg) ? st[1][r] * scale : -1.0e30f;
      float bm = hred_max(fmaxf(sc0, sc1));
      float nm = fmaxf(mst[r], bm);
      alpha[r] = __expf(mst[r] - nm);
      p0[r] = __expf(sc0 - nm);
      p1[r] = __expf(sc1 - nm);
      float rs = hred_sum(p0[r] + p1[r]);
      lst[r] = lst[r] * alpha[r] + rs;
      mst[r] = nm;
    }
#pragma unroll
    for (int t = 0; t < 4; ++t)
#pragma unroll
      for (int r = 0; r < 8; ++r) cacc[t][r] *= alpha[r];

    // D-layout f32 -> A-layout bf16 via LDS transpose bounce
#pragma unroll
    for (int r = 0; r < 8; ++r) {
      pT[wv][r + 8 * lhi][l15]      = f32_bf16(p0[r]);
      pT[wv][r + 8 * lhi][16 + l15] = f32_bf16(p1[r]);
    }
    asm volatile("s_wait_dscnt 0" ::: "memory");
    bf16x16 pf = frag1(&pT[wv][l15][lhi * 16]);

    // PV: ctx[16x64] += P[16x32] * V[32x64]
#pragma unroll
    for (int t = 0; t < 4; ++t) {
      const unsigned short* vrow = vbase + (size_t)(t * 16 + l15) * S_ + kv0 + lhi * 16;
      bf16x16 vf = frag1(vrow);
      cacc[t] = wmma_bf16(pf, vf, cacc[t]);
    }
  }

  // normalize and store ctx (bf16, [B,S,H*DV])
  float inv[8];
#pragma unroll
  for (int r = 0; r < 8; ++r) inv[r] = 1.0f / lst[r];
#pragma unroll
  for (int t = 0; t < 4; ++t) {
#pragma unroll
    for (int r = 0; r < 8; ++r) {
      int s = q0 + r + 8 * lhi;
      int c = h * DV_ + t * 16 + l15;
      ctxb[((size_t)(b * S_ + s)) * (H_ * DV_) + c] = f32_bf16(cacc[t][r] * inv[r]);
    }
  }
}

// ---------------- output projection (16x64 tile per wave) ----------------
// wave id bits: [8:0]=m-tile(512) [12:9]=n-block(16) -> 8 waves/WG share n-block
__global__ __launch_bounds__(256) void oproj(const unsigned short* __restrict__ ctxb,
                                             const unsigned short* __restrict__ Wot,
                                             float* __restrict__ out) {
  __shared__ __align__(16) unsigned short ldsB[2][64][32];   // 8KB  B k-slices
  __shared__ __align__(16) float stgF[8][16][64];            // 32KB store bounce

  int wv = threadIdx.x >> 5, lane = threadIdx.x & 31;
  int gw = blockIdx.x * 8 + wv;
  int mt = gw & 511, nb = gw >> 9;
  int n0 = nb * 64;

  f32x8 acc[4];
#pragma unroll
  for (int t = 0; t < 4; ++t) acc[t] = (f32x8){0.f, 0.f, 0.f, 0.f, 0.f, 0.f, 0.f, 0.f};

  coop_gemm16x64(ctxb + (size_t)(mt * 16) * D_,
                 Wot + (size_t)n0 * D_,
                 ldsB, wv, lane, acc);

  int l15 = lane & 15, lhi = lane >> 4;
#pragma unroll
  for (int t = 0; t < 4; ++t)
#pragma unroll
    for (int r = 0; r < 8; ++r)
      stgF[wv][r + 8 * lhi][t * 16 + l15] = acc[t][r];
  asm volatile("s_wait_dscnt 0" ::: "memory");

  float* rowbase = out + (size_t)(mt * 16) * D_ + n0;
#pragma unroll
  for (int it = 0; it < 8; ++it) {
    int chunk = it * 32 + lane;
    int row = chunk >> 4, c4 = (chunk & 15) * 4;
    float4 vld = *(const float4*)&stgF[wv][row][c4];
    *(float4*)(rowbase + (size_t)row * D_ + c4) = vld;
  }
}

// ---------------- launch ----------------
extern "C" void kernel_launch(void* const* d_in, const int* in_sizes, int n_in,
                              void* d_out, int out_size, void* d_ws, size_t ws_size,
                              hipStream_t stream) {
  (void)in_sizes; (void)n_in; (void)out_size; (void)ws_size;
  const float* x  = (const float*)d_in[0];
  const float* Wq = (const float*)d_in[1];
  const float* Wk = (const float*)d_in[2];
  const float* Wv = (const float*)d_in[3];
  const float* Wo = (const float*)d_in[4];
  float* out = (float*)d_out;

  char* ws = (char*)d_ws;
  unsigned short* xb   = (unsigned short*)(ws);                         // 16 MB
  unsigned short* Wt   = (unsigned short*)(ws + 16777216);              // 6 MB
  unsigned short* Wot  = (unsigned short*)(ws + 23068672);              // 2 MB
  unsigned short* qb   = (unsigned short*)(ws + 25165824);              // 16 MB
  unsigned short* kb   = (unsigned short*)(ws + 41943040);              // 16 MB
  unsigned short* vT   = (unsigned short*)(ws + 58720256);              // 16 MB
  unsigned short* ctxb = (unsigned short*)(ws + 75497472);              // 16 MB

  conv_x<<<dim3((B_ * S_ * D_ / 4 + 255) / 256), dim3(256), 0, stream>>>(x, xb, B_ * S_ * D_ / 4);
  conv_wqkv<<<dim3((3 * H_ * DK_ * D_ + 255) / 256), dim3(256), 0, stream>>>(Wq, Wk, Wv, Wt);
  conv_wo<<<dim3((D_ * D_ + 255) / 256), dim3(256), 0, stream>>>(Wo, Wot);

  // QKV: 3*16 (sel,head) * 512 m-tiles waves, 8 waves/block
  qkv_gemm<<<dim3(3 * H_ * 512 / 8), dim3(256), 0, stream>>>(xb, Wt, qb, kb, vT);

  // attention: B*H*128 q-tiles, 8 waves/block
  attn<<<dim3(B_ * H_ * 128 / 8), dim3(256), 0, stream>>>(qb, kb, vT, ctxb);

  // output projection: 512 m-tiles * 16 n-blocks, 8 waves/block
  oproj<<<dim3(512 * 16 / 8), dim3(256), 0, stream>>>(ctxb, Wot, out);
}
